// SpatialTransformerLayer_21655225106558
// MI455X (gfx1250) — matrix operand order repaired and checked by reading, not compiled, not run on hardware
//
#include <hip/hip_runtime.h>

typedef _Float16 h16;
typedef __attribute__((ext_vector_type(16))) _Float16 v16h;
typedef __attribute__((ext_vector_type(8)))  _Float16 v8h;
typedef __attribute__((ext_vector_type(8)))  float    v8f;
typedef __attribute__((ext_vector_type(8)))  unsigned int v8u;

// ---------------- geometry constants ----------------
#define NB      64
#define IH      224
#define IW      224
#define IC      3
#define C1OH    218      // 224-7+1
#define C1OW    218
#define C1OC    8
#define P1HH    109
#define P1WW    109
#define C2OH    105      // 109-5+1
#define C2OW    105
#define C2OC    10
#define P2HH    52
#define P2WW    52
#define FLATK   27040    // 52*52*10  (= 845*32 exactly)
#define DENSE1N 32

#define P1ELEMS (NB * P1HH * P1WW * C1OC)   // 6,083,072
#define P2PIX   (NB * P2HH * P2WW)          // 173,056

// A-fragment K index (16-bit data, 16x32 tile), within one 32-wide chunk:
//   lanes 0-15: i<8 -> K=i,    i>=8 -> K=8+i   (covers 0..7, 16..23)
//   lanes16-31: i<8 -> K=8+i,  i>=8 -> K=16+i  (covers 8..15, 24..31)
// Pair index pi = K>>1; for pair p (halves i=2p,2p+1):
//   pi = kc*16 + (lane>>4)*4 + p + (p>=4 ? 4 : 0)

// ---------------- weight fragment packing ----------------
// B-fragment layout (16-bit, 32x16): lane l holds N = l&15, K = (l>>4)*8 + i + (i>=8 ? 8 : 0).
// conv1 uses a channel-padded K ordering: K' = (kh*7+kw)*4 + c, c in [0,4), c==3 padded.
__global__ __launch_bounds__(256)
void stn_pack_weights(const float* __restrict__ c1w,   // [7][7][3][8] flat
                      const float* __restrict__ c2w,   // [200][10]
                      const float* __restrict__ d1w,   // [27040][32]
                      h16* __restrict__ w1p,           // 7*32*16
                      h16* __restrict__ w2p,           // 7*32*16
                      h16* __restrict__ d1p) {         // 845*2*32*16
  int idx = blockIdx.x * 256 + threadIdx.x;
  if (idx < 7 * 512) {                       // conv1: K' padded to 224
    int i = idx & 15, lane = (idx >> 4) & 31, kc = idx >> 9;
    int K = kc * 32 + ((lane >> 4) << 3) + i + ((i >= 8) ? 8 : 0);
    int N = lane & 15;
    int c = K & 3, q = K >> 2;
    float v = 0.0f;
    if (c < 3 && q < 49 && N < C1OC) v = c1w[(q * 3 + c) * C1OC + N];
    w1p[idx] = (h16)v;
  } else if (idx < 14 * 512) {               // conv2: natural K, padded 200->224
    int j = idx - 7 * 512;
    int i = j & 15, lane = (j >> 4) & 31, kc = j >> 9;
    int K = kc * 32 + ((lane >> 4) << 3) + i + ((i >= 8) ? 8 : 0);
    int N = lane & 15;
    float v = (K < 200 && N < C2OC) ? c2w[K * C2OC + N] : 0.0f;
    w2p[j] = (h16)v;
  } else if (idx < 14 * 512 + 845 * 2 * 512) {   // dense1, 2 N-tiles
    int j = idx - 14 * 512;
    int i = j & 15, lane = (j >> 4) & 31, pair = j >> 9;  // pair = kc*2+nt
    int nt = pair & 1, kc = pair >> 1;
    int K = kc * 32 + ((lane >> 4) << 3) + i + ((i >= 8) ? 8 : 0);             // always < 27040
    int N = nt * 16 + (lane & 15);
    d1p[j] = (h16)d1w[K * DENSE1N + N];
  }
}

// ---------------- conv1: LDS-staged implicit GEMM, fused bias+ReLU+maxpool ----------------
// grid = (4 ow-groups of 64, 218 rows, 64 batches), block = 128 threads (4 waves)
__global__ __launch_bounds__(128)
void stn_conv1_wmma(const float* __restrict__ x, const h16* __restrict__ w1p,
                    const float* __restrict__ c1b, unsigned int* __restrict__ p1raw) {
  __shared__ h16 tile[7 * 70 * 4];   // input patch, 3ch padded to 4, 3920 B
  __shared__ int ptab[112];          // half-offset per K-pair
  const int tid = threadIdx.x;
  const int n   = blockIdx.z;
  const int oh  = blockIdx.y;
  const int cw0 = blockIdx.x * 64;

  // K-pair offset table: pi -> half offset of (kh,kw,c) in tile. Padded pairs -> 0
  // (their packed weights are zero, so the in-bounds garbage contributes nothing).
  for (int t = tid; t < 112; t += 128) {
    int k = t * 2;
    int c = k & 3, q = k >> 2;     // c in {0,2}; pair (c,c+1) contiguous (c==3 is the pad)
    int off = 0;
    if (q < 49) {
      int kw = q % 7, kh = q / 7;
      off = (kh * 70 + kw) * 4 + c;
    }
    ptab[t] = off;
  }
  // stage 7 x 70 pixels (cols clamped; clamped region only feeds guarded tiles)
  for (int p = tid; p < 7 * 70; p += 128) {
    int r = p / 70, cl = p - r * 70;
    int gc = cw0 + cl; if (gc > IW - 1) gc = IW - 1;
    const float* src = x + ((n * IH + (oh + r)) * IW + gc) * IC;
    h16* dst = tile + p * 4;
    dst[0] = (h16)src[0];
    dst[1] = (h16)src[1];
    dst[2] = (h16)src[2];
    dst[3] = (h16)0.0f;
  }
  __syncthreads();

  const int lane = tid & 31;
  const int wv   = tid >> 5;
  const int jl   = wv * 16 + (lane & 15);   // local output pixel 0..63
  const int pib  = (lane >> 4) << 2;

  v8f acc = {};
  for (int kc = 0; kc < 7; ++kc) {
    v8u au;
#pragma unroll
    for (int p = 0; p < 8; ++p) {
      int pi = kc * 16 + pib + p + ((p >= 4) ? 4 : 0);
      au[p] = *(const unsigned*)(tile + jl * 4 + ptab[pi]);   // 2 halves per dword
    }
    v16h a = __builtin_bit_cast(v16h, au);
    v16h b = *(const v16h*)(w1p + (kc * 32 + lane) * 16);
    acc = __builtin_amdgcn_wmma_f32_16x16x32_f16(false, a, false, b,
                                                 (short)0, acc, false, false);
  }

  const int ch = lane & 15;
  if (ch < C1OC) {
    const float bias = c1b[ch];
#pragma unroll
    for (int r = 0; r < 8; ++r) {
      int jj = wv * 16 + r + ((lane >> 4) << 3);
      int ow = cw0 + jj;
      if (ow < C1OW) {
        float v = acc[r] + bias;
        v = v > 0.0f ? v : 0.0f;               // ReLU (>=0 => uint order == float order)
        unsigned int* dst = p1raw +
            (((n * P1HH + (oh >> 1)) * P1WW + (ow >> 1)) * C1OC + ch);
        atomicMax(dst, __float_as_uint(v));    // fused 2x2 maxpool
      }
    }
  }
}

// ---------------- BN after pool (scale>0 => commutes with max), emit f16 ----------------
__global__ __launch_bounds__(256)
void stn_bn1_f16(const unsigned int* __restrict__ p1raw,
                 const float* __restrict__ g, const float* __restrict__ b,
                 const float* __restrict__ m, const float* __restrict__ v,
                 h16* __restrict__ p1h) {
  int idx = blockIdx.x * 256 + threadIdx.x;
  if (idx >= P1ELEMS) return;
  int ch = idx & 7;
  float s = g[ch] * rsqrtf(v[ch] + 1e-3f);
  float t = b[ch] - m[ch] * s;
  p1h[idx] = (h16)(__uint_as_float(p1raw[idx]) * s + t);
}

// ---------------- conv2: LDS-staged implicit GEMM, fused bias+ReLU+maxpool ----------------
// grid = (2 ow-groups of 64, 105 rows, 64 batches), block = 128 threads
__global__ __launch_bounds__(128)
void stn_conv2_wmma(const h16* __restrict__ p1h, const h16* __restrict__ w2p,
                    const float* __restrict__ c2b, unsigned int* __restrict__ p2raw) {
  __shared__ h16 tile[5 * 68 * 8];   // 5440 B
  __shared__ int ptab[112];
  const int tid = threadIdx.x;
  const int n   = blockIdx.z;
  const int oh  = blockIdx.y;
  const int cw0 = blockIdx.x * 64;

  for (int t = tid; t < 112; t += 128) {
    int k = t * 2;
    int c = k & 7, q = k >> 3;     // c even -> pair (c,c+1) contiguous
    int off = 0;
    if (q < 25) {
      int kw = q % 5, kh = q / 5;
      off = (kh * 68 + kw) * 8 + c;
    }
    ptab[t] = off;
  }
  // stage rows oh..oh+4, cols cw0..cw0+67 (clamped), 8 ch -- dword copies
  for (int d = tid; d < 5 * 68 * 4; d += 128) {
    int r = d / 272, rd = d - r * 272;
    int cl = rd >> 2, cp = rd & 3;
    int gc = cw0 + cl; if (gc > P1WW - 1) gc = P1WW - 1;
    const unsigned* src =
        (const unsigned*)(p1h + ((n * P1HH + (oh + r)) * P1WW + gc) * C1OC) + cp;
    ((unsigned*)tile)[d] = *src;
  }
  __syncthreads();

  const int lane = tid & 31;
  const int wv   = tid >> 5;
  const int jl   = wv * 16 + (lane & 15);
  const int pib  = (lane >> 4) << 2;

  v8f acc = {};
  for (int kc = 0; kc < 7; ++kc) {
    v8u au;
#pragma unroll
    for (int p = 0; p < 8; ++p) {
      int pi = kc * 16 + pib + p + ((p >= 4) ? 4 : 0);
      au[p] = *(const unsigned*)(tile + jl * 8 + ptab[pi]);
    }
    v16h a = __builtin_bit_cast(v16h, au);
    v16h b = *(const v16h*)(w2p + (kc * 32 + lane) * 16);
    acc = __builtin_amdgcn_wmma_f32_16x16x32_f16(false, a, false, b,
                                                 (short)0, acc, false, false);
  }

  const int ch = lane & 15;
  if (ch < C2OC) {
    const float bias = c2b[ch];
#pragma unroll
    for (int r = 0; r < 8; ++r) {
      int jj = wv * 16 + r + ((lane >> 4) << 3);
      int ow = cw0 + jj;
      if (ow < 104 && oh < 104) {              // 105 odd: row/col 104 dropped by pool
        float v = acc[r] + bias;
        v = v > 0.0f ? v : 0.0f;
        unsigned int* dst = p2raw +
            (((n * P2HH + (oh >> 1)) * P2WW + (ow >> 1)) * C2OC + ch);
        atomicMax(dst, __float_as_uint(v));
      }
    }
  }
}

__global__ __launch_bounds__(256)
void stn_bn2_f16(const unsigned int* __restrict__ p2raw,
                 const float* __restrict__ g, const float* __restrict__ b,
                 const float* __restrict__ m, const float* __restrict__ v,
                 h16* __restrict__ p2h) {
  int pix = blockIdx.x * 256 + threadIdx.x;    // one thread per pooled pixel
  if (pix >= P2PIX) return;
#pragma unroll
  for (int c = 0; c < C2OC; ++c) {
    float s = g[c] * rsqrtf(v[c] + 1e-3f);
    float t = b[c] - m[c] * s;
    int idx = pix * C2OC + c;
    p2h[idx] = (h16)(__uint_as_float(p2raw[idx]) * s + t);
  }
}

// ---------------- dense1: [64 x 27040] x [27040 x 32], K split 65 ways ----------------
__global__ __launch_bounds__(128)
void stn_dense1_wmma(const h16* __restrict__ p2h, const h16* __restrict__ d1p,
                     float* __restrict__ h3) {
  const int lane = threadIdx.x & 31;
  const int wid  = blockIdx.x * 4 + (threadIdx.x >> 5);   // 0..519
  const int ks   = wid % 65;
  const int nt   = (wid / 65) & 1;
  const int mt   = wid / 130;
  const int row  = mt * 16 + (lane & 15);                 // batch index
  const int kb   = (lane >> 4) << 3;

  v8f acc = {};
  for (int j = 0; j < 13; ++j) {
    int kc = ks * 13 + j;
    const h16* ap = p2h + row * FLATK + kc * 32 + kb;     // two contiguous 8-half runs
    v8h lo = *(const v8h*)(ap);
    v8h hi = *(const v8h*)(ap + 16);
    v16h a;
#pragma unroll
    for (int i = 0; i < 8; ++i) { a[i] = lo[i]; a[i + 8] = hi[i]; }
    v16h b = *(const v16h*)(d1p + ((kc * 2 + nt) * 32 + lane) * 16);
    acc = __builtin_amdgcn_wmma_f32_16x16x32_f16(false, a, false, b,
                                                 (short)0, acc, false, false);
  }
#pragma unroll
  for (int r = 0; r < 8; ++r) {
    int m = mt * 16 + r + ((lane >> 4) << 3);
    int nidx = nt * 16 + (lane & 15);
    atomicAdd(&h3[m * DENSE1N + nidx], acc[r]);
  }
}

// ---------------- dense2: theta = relu(h3+b1) @ d2_w + d2_b ----------------
__global__ __launch_bounds__(64)
void stn_dense2_theta(const float* __restrict__ h3, const float* __restrict__ d1b,
                      const float* __restrict__ d2w, const float* __restrict__ d2b,
                      float* __restrict__ theta) {
  int n = threadIdx.x;
  if (n >= NB) return;
  float h[32];
#pragma unroll
  for (int j = 0; j < 32; ++j) {
    float v = h3[n * 32 + j] + d1b[j];
    h[j] = v > 0.0f ? v : 0.0f;
  }
#pragma unroll
  for (int o = 0; o < 6; ++o) {
    float s = d2b[o];
#pragma unroll
    for (int j = 0; j < 32; ++j) s += h[j] * d2w[j * 6 + o];
    theta[n * 6 + o] = s;
  }
}

// ---------------- bilinear sampler ----------------
__device__ __forceinline__ float stn_tap(const float* img, int yi, int xi, int c) {
  bool valid = (yi >= 0) & (yi < IH) & (xi >= 0) & (xi < IW);
  int yc = min(max(yi, 0), IH - 1);
  int xc = min(max(xi, 0), IW - 1);
  float v = img[(yc * IW + xc) * IC + c];
  return valid ? v : 0.0f;
}

// grid = (IH, NB), block = IW threads: (n, oh, ow) straight from IDs, no divisions
__global__ __launch_bounds__(IW)
void stn_sample(const float* __restrict__ x, const float* __restrict__ theta,
                float* __restrict__ out) {
  const int ow = threadIdx.x;
  const int oh = blockIdx.x;
  const int n  = blockIdx.y;

  const float* t = theta + n * 6;
  float xg = ow * (2.0f / 223.0f) - 1.0f;
  float yg = oh * (2.0f / 223.0f) - 1.0f;
  float cx = t[0] * xg + t[1] * yg + t[2];
  float cy = t[3] * xg + t[4] * yg + t[5];
  float px = (cx + 1.0f) * 0.5f * 223.0f;
  float py = (cy + 1.0f) * 0.5f * 223.0f;

  int x0 = (int)floorf(px), y0 = (int)floorf(py);
  float wx = px - (float)x0, wy = py - (float)y0;
  float w00 = (1.0f - wx) * (1.0f - wy);
  float w01 = wx * (1.0f - wy);
  float w10 = (1.0f - wx) * wy;
  float w11 = wx * wy;

  const float* img = x + n * IH * IW * IC;
  const int obase = ((n * IH + oh) * IW + ow) * IC;
#pragma unroll
  for (int c = 0; c < IC; ++c) {
    float v = stn_tap(img, y0,     x0,     c) * w00
            + stn_tap(img, y0,     x0 + 1, c) * w01
            + stn_tap(img, y0 + 1, x0,     c) * w10
            + stn_tap(img, y0 + 1, x0 + 1, c) * w11;
    out[obase + c] = v;
  }
}

// ---------------- host launcher ----------------
extern "C" void kernel_launch(void* const* d_in, const int* in_sizes, int n_in,
                              void* d_out, int out_size, void* d_ws, size_t ws_size,
                              hipStream_t stream) {
  const float* x   = (const float*)d_in[0];
  const float* c1w = (const float*)d_in[1];
  const float* c1b = (const float*)d_in[2];
  const float* g1  = (const float*)d_in[3];
  const float* b1  = (const float*)d_in[4];
  const float* m1  = (const float*)d_in[5];
  const float* v1  = (const float*)d_in[6];
  const float* c2w = (const float*)d_in[7];
  const float* c2b = (const float*)d_in[8];
  const float* g2  = (const float*)d_in[9];
  const float* b2  = (const float*)d_in[10];
  const float* m2  = (const float*)d_in[11];
  const float* v2  = (const float*)d_in[12];
  const float* d1w = (const float*)d_in[13];
  const float* d1b = (const float*)d_in[14];
  const float* d2w = (const float*)d_in[15];
  const float* d2b = (const float*)d_in[16];
  float* out = (float*)d_out;

  char* ws = (char*)d_ws;
  unsigned int* p1raw = (unsigned int*)(ws + 0);          // 24,332,288 B
  h16*   p1h  = (h16*)(ws + 24332288);                    // 12,166,144 B
  unsigned int* p2raw = (unsigned int*)(ws + 36498432);   //  6,922,240 B
  h16*   p2h  = (h16*)(ws + 43420672);                    //  3,461,120 B
  h16*   w1p  = (h16*)(ws + 46881792);                    //      7,168 B
  h16*   w2p  = (h16*)(ws + 46888960);                    //      7,168 B
  h16*   d1p  = (h16*)(ws + 46896128);                    //  1,730,560 B
  float* h3   = (float*)(ws + 48626688);                  //      8,192 B
  float* th   = (float*)(ws + 48634880);                  //      1,536 B

  // zero atomic-reduction targets (graph-capturable memset nodes)
  hipMemsetAsync(p1raw, 0, 24332288, stream);
  hipMemsetAsync(p2raw, 0, 6922240, stream);
  hipMemsetAsync(h3,    0, 8192,    stream);

  stn_pack_weights<<<3408, 256, 0, stream>>>(c1w, c2w, d1w, w1p, w2p, d1p);
  stn_conv1_wmma <<<dim3(4, C1OH, NB), 128, 0, stream>>>(x, w1p, c1b, p1raw);
  stn_bn1_f16    <<<P1ELEMS / 256, 256, 0, stream>>>(p1raw, g1, b1, m1, v1, p1h);
  stn_conv2_wmma <<<dim3(2, C2OH, NB), 128, 0, stream>>>(p1h, w2p, c2b, p2raw);
  stn_bn2_f16    <<<(P2PIX + 255) / 256, 256, 0, stream>>>(p2raw, g2, b2, m2, v2, p2h);
  stn_dense1_wmma<<<130, 128, 0, stream>>>(p2h, d1p, h3);                 // 520 waves
  stn_dense2_theta<<<1, 64, 0, stream>>>(h3, d1b, d2w, d2b, th);
  stn_sample     <<<dim3(IH, NB), IW, 0, stream>>>(x, th, out);
}
